// SFGCN_19164144075577
// MI455X (gfx1250) — matrix operand order repaired and checked by reading, not compiled, not run on hardware
//
#include <hip/hip_runtime.h>
#include <hip/hip_bf16.h>

#define NFEAT 128
#define NHID1 128
#define NHID2 64

typedef __attribute__((ext_vector_type(16))) __bf16       v16bf;
typedef __attribute__((ext_vector_type(8)))  float        v8f;
typedef __attribute__((ext_vector_type(8)))  unsigned int v8u;

union FragU { v8u u; v16bf b; };

__device__ __forceinline__ unsigned short f2bf(float f) {
    unsigned int u = __float_as_uint(f);
    unsigned int r = (u + 0x7FFFu + ((u >> 16) & 1u)) >> 16;   // RNE
    return (unsigned short)r;
}

// ---------------- elementwise: fp32 -> bf16 ----------------
__global__ void cast_bf16_kernel(const float* __restrict__ in,
                                 unsigned short* __restrict__ out, long n) {
    long i = (long)blockIdx.x * blockDim.x + threadIdx.x;
    if (i < n) out[i] = f2bf(in[i]);
}

// ---------------- transpose + cast: W[K][Nout] fp32 -> Wt[Nout][K] bf16 ----
__global__ void transpose_cast_kernel(const float* __restrict__ W,
                                      unsigned short* __restrict__ Wt,
                                      int K, int Nout) {
    int i = blockIdx.x * blockDim.x + threadIdx.x;
    if (i >= K * Nout) return;
    int k = i / Nout, n = i % Nout;
    Wt[n * K + k] = f2bf(W[k * Nout + n]);
}

// ---------------- bias + relu + cast: H fp32 (N x F) -> Hb bf16 -----------
__global__ void bias_relu_cast_kernel(const float* __restrict__ H,
                                      const float* __restrict__ bias,
                                      unsigned short* __restrict__ Hb,
                                      long n, int F) {
    long i = (long)blockIdx.x * blockDim.x + threadIdx.x;
    if (i >= n) return;
    int f = (int)(i % F);
    float v = H[i] + bias[f];
    Hb[i] = f2bf(v > 0.0f ? v : 0.0f);
}

// ---------------- WMMA bf16 GEMM: O[z] = A[z] (nrows x 128) @ Bt[sel]^T ----
// A row-major bf16 (K=128), Bt row-major bf16 [Nout][128], O fp32 row-major.
// Bt tile (<=32KB) staged in LDS once per block; fragments fed by ds_load_b128.
__global__ __launch_bounds__(256)
void gemm_bf16_wmma(const unsigned short* __restrict__ A,
                    const unsigned short* __restrict__ Bt,
                    float* __restrict__ O,
                    int nrows, int Nout,
                    long a_stride, long b_stride, long o_stride,
                    unsigned int b_map) {
    const int K = 128;
    __shared__ unsigned short sB[128 * 128];          // 32 KB max

    int z = blockIdx.z;
    A  += (long)z * a_stride;
    Bt += (long)((b_map >> (4 * z)) & 0xF) * b_stride;
    O  += (long)z * o_stride;

    // Cooperative stage of the whole weight tile into LDS (before any wave exits).
    {
        const v8u* src = (const v8u*)Bt;               // 32B vectors
        v8u* dst = (v8u*)sB;
        int nvec = (Nout * K) >> 4;                    // ushorts / 16
        for (int i = threadIdx.x; i < nvec; i += blockDim.x) dst[i] = src[i];
    }
    __syncthreads();

    // Scalarize the wave index so tile / m0 / full are SGPRs (scalar branches,
    // no exec-mask dance around the store path).
    int wv   = __builtin_amdgcn_readfirstlane(threadIdx.x >> 5);
    int lane = threadIdx.x & 31;
    int tile = blockIdx.x * 8 + wv;
    int m0   = tile * 16;
    if (m0 >= nrows) return;                           // scalar branch

    int half = lane >> 4;
    int ml   = lane & 15;
    int row  = m0 + ml; if (row >= nrows) row = nrows - 1;

    // Preload 4 A fragments (K-steps of 32). ISA layout: lanes 0-15 hold
    // K pairs {0..7,16..23}, lanes 16-31 hold {8..15,24..31} (+ks*32).
    FragU afr[4];
#pragma unroll
    for (int ks = 0; ks < 4; ++ks) {
#pragma unroll
        for (int v = 0; v < 8; ++v) {
            int kb = ks * 32 + ((v < 4) ? 2 * v : 16 + 2 * (v - 4)) + 8 * half;
            afr[ks].u[v] = *(const unsigned int*)(A + (long)row * K + kb);
        }
    }

    bool full = (m0 + 16) <= nrows;                    // scalar
    int ntiles = Nout >> 4;
    for (int nt = 0; nt < ntiles; ++nt) {
        int n = nt * 16 + ml;                          // column owned by this lane
        v8f acc = {};
#pragma unroll
        for (int ks = 0; ks < 4; ++ks) {
            FragU bfr;
#pragma unroll
            for (int v = 0; v < 8; ++v) {
                int kb = ks * 32 + ((v < 4) ? 2 * v : 16 + 2 * (v - 4)) + 8 * half;
                bfr.u[v] = *(const unsigned int*)(sB + n * K + kb);
            }
            acc = __builtin_amdgcn_wmma_f32_16x16x32_bf16(
                false, afr[ks].b, false, bfr.b, (short)0, acc, false, false);
        }
        // C layout: VGPR r -> M = r + 8*half, N = ml
        float* obase = O + (long)(m0 + 8 * half) * Nout + nt * 16 + ml;
        if (full) {
#pragma unroll
            for (int r = 0; r < 8; ++r) obase[(long)r * Nout] = acc[r];
        } else {
#pragma unroll
            for (int r = 0; r < 8; ++r)
                if (m0 + 8 * half + r < nrows) obase[(long)r * Nout] = acc[r];
        }
    }
}

// ---------------- edge-parallel SpMM scatter: Out[row] += val * H[col] ----
// One wave32 per edge; scalar (SMEM) loads of the wave-uniform edge indices.
__global__ void spmm_atomic_kernel(const int* __restrict__ row,
                                   const int* __restrict__ col,
                                   const float* __restrict__ val,
                                   const float* __restrict__ H,
                                   float* __restrict__ Out,
                                   long E, int F) {
    int wv = __builtin_amdgcn_readfirstlane(threadIdx.x >> 5);
    long e = (long)blockIdx.x * 8 + wv;                // scalar
    int lane = threadIdx.x & 31;
    if (e >= E) return;                                // scalar branch
    int r = row[e], c = col[e];                        // s_load
    float v = val[e];
    const float* hp = H + (long)c * F;
    float* op = Out + (long)r * F;
    for (int f = lane; f < F; f += 32)
        atomicAdd(&op[f], v * hp[f]);
}

// ---------------- fused head: biases, attention, softmax, MLP, outputs ----
__global__ __launch_bounds__(256)
void head_kernel(const float* __restrict__ s2o,      // 4 x (N x 64), contiguous
                 const float* __restrict__ b_s1_2,
                 const float* __restrict__ b_c_2,
                 const float* __restrict__ b_s2_2,
                 const float* __restrict__ att_w1,   // 64 x 16
                 const float* __restrict__ att_b1,   // 16
                 const float* __restrict__ att_w2,   // 16
                 const float* __restrict__ mlp_w,    // 64 x 8
                 const float* __restrict__ mlp_b,    // 8
                 float* __restrict__ out, int N) {
    __shared__ float sW1[64 * 16];
    __shared__ float sB1[16];
    __shared__ float sW2[16];
    __shared__ float sMW[64 * 8];
    __shared__ float sMB[8];
    __shared__ float sBs1[64], sBc[64], sBs2[64];
    __shared__ float zbuf[8][256];            // per wave: 3x64 z, 64 emb

    int tid = threadIdx.x;
    for (int i = tid; i < 1024; i += 256) sW1[i] = att_w1[i];
    for (int i = tid; i < 512;  i += 256) sMW[i] = mlp_w[i];
    for (int i = tid; i < 64;   i += 256) { sBs1[i] = b_s1_2[i]; sBc[i] = b_c_2[i]; sBs2[i] = b_s2_2[i]; }
    if (tid < 16) { sB1[tid] = att_b1[tid]; sW2[tid] = att_w2[tid]; }
    if (tid < 8)  sMB[tid] = mlp_b[tid];
    __syncthreads();

    int wv = __builtin_amdgcn_readfirstlane(tid >> 5);
    int lane = tid & 31;
    int node = blockIdx.x * 8 + wv;                   // scalar
    bool valid = node < N;                            // scalar
    int nidx = valid ? node : (N - 1);
    long base = (long)nidx * 64;
    long NL = N;
    const float* p0 = s2o;
    const float* p1 = s2o + NL * 64;
    const float* p2 = s2o + NL * 128;
    const float* p3 = s2o + NL * 192;

    int d0 = lane, d1 = lane + 32;
    float e1a = p0[base + d0] + sBs1[d0], e1b = p0[base + d1] + sBs1[d1];
    float c1a = p1[base + d0] + sBc[d0],  c1b = p1[base + d1] + sBc[d1];
    float c2a = p2[base + d0] + sBc[d0],  c2b = p2[base + d1] + sBc[d1];
    float e2a = p3[base + d0] + sBs2[d0], e2b = p3[base + d1] + sBs2[d1];
    float xca = 0.5f * (c1a + c2a), xcb = 0.5f * (c1b + c2b);

    // raw output offsets: output[N,8], beta[N,3], emb1/com1/com2/emb2/emb [N,64]
    long o_beta = NL * 8, o_e1 = NL * 11, o_c1 = NL * 75,
         o_c2 = NL * 139, o_e2 = NL * 203, o_emb = NL * 267;
    if (valid) {
        long nb = (long)node * 64;
        out[o_e1 + nb + d0] = e1a; out[o_e1 + nb + d1] = e1b;
        out[o_c1 + nb + d0] = c1a; out[o_c1 + nb + d1] = c1b;
        out[o_c2 + nb + d0] = c2a; out[o_c2 + nb + d1] = c2b;
        out[o_e2 + nb + d0] = e2a; out[o_e2 + nb + d1] = e2b;
    }
    // z = stack([emb1, emb2, xcom])
    zbuf[wv][0   + d0] = e1a; zbuf[wv][0   + d1] = e1b;
    zbuf[wv][64  + d0] = e2a; zbuf[wv][64  + d1] = e2b;
    zbuf[wv][128 + d0] = xca; zbuf[wv][128 + d1] = xcb;
    __syncthreads();

    // attention scores: w_p = tanh(z_p @ W1 + b1) @ W2
    float wp[3];
#pragma unroll
    for (int p = 0; p < 3; ++p) {
        float t = 0.0f;
        if (lane < 16) {
            float acc = 0.0f;
            for (int d = 0; d < 64; ++d)
                acc += zbuf[wv][p * 64 + d] * sW1[d * 16 + lane];
            t = tanhf(acc + sB1[lane]) * sW2[lane];
        }
#pragma unroll
        for (int off = 16; off >= 1; off >>= 1) t += __shfl_xor(t, off, 32);
        wp[p] = t;
    }
    float m = fmaxf(wp[0], fmaxf(wp[1], wp[2]));
    float x0 = expf(wp[0] - m), x1 = expf(wp[1] - m), x2 = expf(wp[2] - m);
    float inv = 1.0f / (x0 + x1 + x2);
    float b0 = x0 * inv, b1 = x1 * inv, b2 = x2 * inv;

    float ema = b0 * e1a + b1 * e2a + b2 * xca;
    float emb = b0 * e1b + b1 * e2b + b2 * xcb;
    zbuf[wv][192 + d0] = ema;
    zbuf[wv][192 + d1] = emb;
    if (valid) {
        long nb = (long)node * 64;
        out[o_emb + nb + d0] = ema; out[o_emb + nb + d1] = emb;
        if (lane < 3) {
            float bb = (lane == 0) ? b0 : (lane == 1) ? b1 : b2;
            out[o_beta + (long)node * 3 + lane] = bb;
        }
    }
    __syncthreads();

    // MLP + log_softmax (lanes 0-7: one class each)
    float acc = 0.0f;
    {
        int k = lane & 7;
        acc = sMB[k];
        for (int d = 0; d < 64; ++d)
            acc += zbuf[wv][192 + d] * sMW[d * 8 + k];
    }
    float mx = acc;
#pragma unroll
    for (int off = 4; off >= 1; off >>= 1) mx = fmaxf(mx, __shfl_xor(mx, off, 32));
    float ex = expf(acc - mx);
    float sm = ex;
#pragma unroll
    for (int off = 4; off >= 1; off >>= 1) sm += __shfl_xor(sm, off, 32);
    float res = acc - mx - logf(sm);
    if (valid && lane < 8) out[(long)node * 8 + lane] = res;
}

extern "C" void kernel_launch(void* const* d_in, const int* in_sizes, int n_in,
                              void* d_out, int out_size, void* d_ws, size_t ws_size,
                              hipStream_t stream) {
    const float* x        = (const float*)d_in[0];
    const int*   srow     = (const int*)d_in[1];
    const int*   scol     = (const int*)d_in[2];
    const float* sval     = (const float*)d_in[3];
    const int*   frow     = (const int*)d_in[4];
    const int*   fcol     = (const int*)d_in[5];
    const float* fval     = (const float*)d_in[6];
    const float* w_s1_1 = (const float*)d_in[7];  const float* b_s1_1 = (const float*)d_in[8];
    const float* w_s1_2 = (const float*)d_in[9];  const float* b_s1_2 = (const float*)d_in[10];
    const float* w_s2_1 = (const float*)d_in[11]; const float* b_s2_1 = (const float*)d_in[12];
    const float* w_s2_2 = (const float*)d_in[13]; const float* b_s2_2 = (const float*)d_in[14];
    const float* w_c_1  = (const float*)d_in[15]; const float* b_c_1  = (const float*)d_in[16];
    const float* w_c_2  = (const float*)d_in[17]; const float* b_c_2  = (const float*)d_in[18];
    const float* att_w1 = (const float*)d_in[19]; const float* att_b1 = (const float*)d_in[20];
    const float* att_w2 = (const float*)d_in[21];
    const float* mlp_w  = (const float*)d_in[22]; const float* mlp_b  = (const float*)d_in[23];

    const long N = in_sizes[0] / NFEAT;
    const long E = in_sizes[1];
    float* out = (float*)d_out;

    // ---- workspace layout (regions reused across stages; peak ~384 MB) ----
    char* w = (char*)d_ws;
    size_t off = 0;
    auto al = [](size_t o) { return (o + 255) & ~(size_t)255; };
    unsigned short* xb  = (unsigned short*)(w + off); off = al(off + (size_t)N * 128 * 2);
    unsigned short* wt1 = (unsigned short*)(w + off); off = al(off + (size_t)3 * 128 * 128 * 2);
    unsigned short* wt2 = (unsigned short*)(w + off); off = al(off + (size_t)3 * 128 * 64 * 2);
    float* xw = (float*)(w + off); off = al(off + (size_t)3 * N * 128 * 4);  // gemm1 out
    float* h1 = (float*)(w + off); off = al(off + (size_t)4 * N * 128 * 4);  // spmm1 out
    unsigned short* hb = (unsigned short*)xw;         // reuse (xw dead after spmm1)
    float* g2  = h1;                                  // reuse (h1 dead after cast)
    float* s2o = h1 + (size_t)4 * N * 64;

    const int T = 256;

    // Stage 1: x -> bf16
    {
        long n = N * 128;
        cast_bf16_kernel<<<dim3((n + T - 1) / T), dim3(T), 0, stream>>>(x, xb, n);
    }
    // Stage 2: transpose+cast weights (wt1: [s1_1, s2_1, c_1]; wt2: [s1_2, c_2, s2_2])
    {
        int n1 = 128 * 128, n2 = 128 * 64;
        transpose_cast_kernel<<<dim3((n1 + T - 1) / T), dim3(T), 0, stream>>>(w_s1_1, wt1,              128, 128);
        transpose_cast_kernel<<<dim3((n1 + T - 1) / T), dim3(T), 0, stream>>>(w_s2_1, wt1 + n1,         128, 128);
        transpose_cast_kernel<<<dim3((n1 + T - 1) / T), dim3(T), 0, stream>>>(w_c_1,  wt1 + 2 * n1,     128, 128);
        transpose_cast_kernel<<<dim3((n2 + T - 1) / T), dim3(T), 0, stream>>>(w_s1_2, wt2,              128, 64);
        transpose_cast_kernel<<<dim3((n2 + T - 1) / T), dim3(T), 0, stream>>>(w_c_2,  wt2 + n2,         128, 64);
        transpose_cast_kernel<<<dim3((n2 + T - 1) / T), dim3(T), 0, stream>>>(w_s2_2, wt2 + 2 * n2,     128, 64);
    }
    // Stage 3: GEMM1 -> xw[z]: z0=x@w_s1_1, z1=x@w_s2_1, z2=x@w_c_1
    {
        long tiles = (N + 15) / 16;
        dim3 grid((unsigned)((tiles + 7) / 8), 1, 3);
        gemm_bf16_wmma<<<grid, dim3(T), 0, stream>>>(
            xb, wt1, xw, (int)N, 128, 0L, (long)128 * 128, N * 128, 0x210u);
    }
    // Stage 4: zero + SpMM1 (paths: emb1|sadj|xw_s1, com1|sadj|xw_c, com2|fadj|xw_c, emb2|fadj|xw_s2)
    hipMemsetAsync(h1, 0, (size_t)4 * N * 128 * 4, stream);
    {
        long thr = E * 32;
        dim3 grid((unsigned)((thr + T - 1) / T));
        spmm_atomic_kernel<<<grid, dim3(T), 0, stream>>>(srow, scol, sval, xw,               h1,               E, 128);
        spmm_atomic_kernel<<<grid, dim3(T), 0, stream>>>(srow, scol, sval, xw + 2 * N * 128, h1 + N * 128,     E, 128);
        spmm_atomic_kernel<<<grid, dim3(T), 0, stream>>>(frow, fcol, fval, xw + 2 * N * 128, h1 + 2 * N * 128, E, 128);
        spmm_atomic_kernel<<<grid, dim3(T), 0, stream>>>(frow, fcol, fval, xw + 1 * N * 128, h1 + 3 * N * 128, E, 128);
    }
    // Stage 5: bias + relu + cast -> hb
    {
        long n = N * 128;
        dim3 grid((unsigned)((n + T - 1) / T));
        bias_relu_cast_kernel<<<grid, dim3(T), 0, stream>>>(h1,               b_s1_1, hb,               n, 128);
        bias_relu_cast_kernel<<<grid, dim3(T), 0, stream>>>(h1 + N * 128,     b_c_1,  hb + N * 128,     n, 128);
        bias_relu_cast_kernel<<<grid, dim3(T), 0, stream>>>(h1 + 2 * N * 128, b_c_1,  hb + 2 * N * 128, n, 128);
        bias_relu_cast_kernel<<<grid, dim3(T), 0, stream>>>(h1 + 3 * N * 128, b_s2_1, hb + 3 * N * 128, n, 128);
    }
    // Stage 6: GEMM2 -> g2[z] (weights: z0->s1_2, z1->c_2, z2->c_2, z3->s2_2)
    {
        long tiles = (N + 15) / 16;
        dim3 grid((unsigned)((tiles + 7) / 8), 1, 4);
        gemm_bf16_wmma<<<grid, dim3(T), 0, stream>>>(
            hb, wt2, g2, (int)N, 64, N * 128, (long)128 * 64, N * 64, 0x2110u);
    }
    // Stage 7: zero + SpMM2
    hipMemsetAsync(s2o, 0, (size_t)4 * N * 64 * 4, stream);
    {
        long thr = E * 32;
        dim3 grid((unsigned)((thr + T - 1) / T));
        spmm_atomic_kernel<<<grid, dim3(T), 0, stream>>>(srow, scol, sval, g2,              s2o,              E, 64);
        spmm_atomic_kernel<<<grid, dim3(T), 0, stream>>>(srow, scol, sval, g2 + N * 64,     s2o + N * 64,     E, 64);
        spmm_atomic_kernel<<<grid, dim3(T), 0, stream>>>(frow, fcol, fval, g2 + 2 * N * 64, s2o + 2 * N * 64, E, 64);
        spmm_atomic_kernel<<<grid, dim3(T), 0, stream>>>(frow, fcol, fval, g2 + 3 * N * 64, s2o + 3 * N * 64, E, 64);
    }
    // Stage 8: fused attention / softmax / MLP head (wave32 per node)
    {
        dim3 grid((unsigned)((N + 7) / 8));
        head_kernel<<<grid, dim3(T), 0, stream>>>(
            s2o, b_s1_2, b_c_2, b_s2_2, att_w1, att_b1, att_w2, mlp_w, mlp_b,
            out, (int)N);
    }
}
